// StageNet_79860621902249
// MI455X (gfx1250) — matrix operand order repaired and checked by reading, not compiled, not run on hardware
//
#include <hip/hip_runtime.h>
#include <hip/hip_bf16.h>

// ---------------------------------------------------------------------------
// StageNet forward, MI455X (gfx1250, wave32, WMMA bf16).
//
//  * xi @ Wk.T hoisted out of the scan -> one 12800x1552x8192 GEMM (bf16 WMMA)
//  * per-step recurrent GEMM h @ Wr.T (256x1552x384) also bf16 WMMA
//  * interval column handled as rank-1 epilogue (colI = Wk[:,ND] + Wr[:,H])
//  * theme/conv/Wo projection executed only at t == last[b]
//  * GEMM k-loops double-buffered (2 fragment sets, k unrolled by 2) so the
//    next-k load clause issues before the current-k WMMA block.
// ---------------------------------------------------------------------------

typedef __attribute__((ext_vector_type(16))) __bf16      v16bf;
typedef __attribute__((ext_vector_type(8)))  float       v8f;
typedef __attribute__((ext_vector_type(4)))  unsigned int v4u;

#define B_   256
#define V_   50
#define N_   64
#define D_   128
#define ND_  8192
#define H_   384
#define L_   8
#define CH_  48
#define KW_  10      // conv window
#define OUT_ 128
#define G_   1552    // 4H + 2L
#define GP_  1664    // G padded to 13*128

// ---------------- helpers ----------------

static __device__ __forceinline__ unsigned short f2bf(float f) {
  unsigned int u = __float_as_uint(f);
  u += 0x7FFFu + ((u >> 16) & 1u);           // round-to-nearest-even
  return (unsigned short)(u >> 16);
}

static __device__ __forceinline__ float sigm(float x) {
  return 1.0f / (1.0f + __expf(-x));
}

// A fragment (16x32 bf16, row-major source, stride ldk):
// lanes 0-15: row m, K {k0..k0+7, k0+16..k0+23}; lanes 16-31: K {+8..+15, +24..+31}
static __device__ __forceinline__ v16bf load_fragA(const unsigned short* __restrict__ base,
                                                   int row, int ldk, int k0, int lane) {
  const unsigned short* p = base + (size_t)row * ldk + (size_t)(k0 + ((lane >> 4) << 3));
  union { v4u u[2]; v16bf b; } f;
  f.u[0] = *(const v4u*)p;
  f.u[1] = *(const v4u*)(p + 16);
  return f.b;
}

// B fragment (32x16 bf16). B is stored transposed (N-major rows of length K),
// so lane n reads row (ncol): lanes 0-15 K {k0..k0+7, k0+16..k0+23},
// lanes 16-31 K {+8..+15, +24..+31}.
static __device__ __forceinline__ v16bf load_fragB(const unsigned short* __restrict__ base,
                                                   int row, int ldk, int k0, int lane) {
  const unsigned short* p = base + (size_t)row * ldk + (size_t)(k0 + ((lane >> 4) << 3));
  union { v4u u[2]; v16bf b; } f;
  f.u[0] = *(const v4u*)p;
  f.u[1] = *(const v4u*)(p + 16);
  return f.b;
}

static __device__ __forceinline__ v8f wmma_bf16(v16bf a, v16bf b, v8f c) {
  return __builtin_amdgcn_wmma_f32_16x16x32_bf16(false, a, false, b, (short)0, c,
                                                 false, false);
}

// ---------------- prep kernels ----------------

// X[row, k] = bf16(embed[node_ids[row, k>>7], k&127]);  row = b*V + v
__global__ void gather_x_kernel(const int* __restrict__ ids,
                                const float* __restrict__ embed,
                                unsigned short* __restrict__ X) {
  int row = blockIdx.x;
  const int* idr = ids + (size_t)row * N_;
  unsigned short* xr = X + (size_t)row * ND_;
  for (int k = threadIdx.x; k < ND_; k += 256) {
    int n = k >> 7, d = k & 127;
    int id = idr[n];
    xr[k] = f2bf(embed[(size_t)id * D_ + d]);
  }
}

// Wkbf [GP_, ND_] bf16, zero-padded rows >= G_
__global__ void pad_wk_kernel(const float* __restrict__ Wk,
                              unsigned short* __restrict__ Wkbf) {
  int g = blockIdx.x;
  unsigned short* wr = Wkbf + (size_t)g * ND_;
  if (g < G_) {
    const float* src = Wk + (size_t)g * (ND_ + 1);
    for (int k = threadIdx.x; k < ND_; k += 256) wr[k] = f2bf(src[k]);
  } else {
    for (int k = threadIdx.x; k < ND_; k += 256) wr[k] = 0;
  }
}

// Wrbf [GP_, H_] bf16, zero-padded
__global__ void pad_wr_kernel(const float* __restrict__ Wr,
                              unsigned short* __restrict__ Wrbf) {
  int g = blockIdx.x;
  unsigned short* wr = Wrbf + (size_t)g * H_;
  if (g < G_) {
    const float* src = Wr + (size_t)g * (H_ + 1);
    for (int k = threadIdx.x; k < H_; k += 128) wr[k] = f2bf(src[k]);
  } else {
    for (int k = threadIdx.x; k < H_; k += 128) wr[k] = 0;
  }
}

// bias1[g] = bk[g] + br[g];  colI[g] = Wk[g, ND] + Wr[g, H]  (interval column)
__global__ void prep_misc_kernel(const float* __restrict__ Wk, const float* __restrict__ Wr,
                                 const float* __restrict__ bk, const float* __restrict__ br,
                                 float* __restrict__ bias1, float* __restrict__ colI) {
  int g = blockIdx.x * 128 + threadIdx.x;
  if (g >= GP_) return;
  if (g < G_) {
    bias1[g] = bk[g] + br[g];
    colI[g]  = Wk[(size_t)g * (ND_ + 1) + ND_] + Wr[(size_t)g * (H_ + 1) + H_];
  } else {
    bias1[g] = 0.f;
    colI[g]  = 0.f;
  }
}

// last[b] from attn_mask (bool bytes): last = count(!all(mask[b,v,:])) - 1 (wrap -1 -> V-1)
__global__ void last_kernel(const unsigned char* __restrict__ mask, int* __restrict__ last) {
  int b = threadIdx.x;
  if (b >= B_) return;
  int cnt = 0;
  for (int v = 0; v < V_; ++v) {
    const unsigned char* mp = mask + ((size_t)b * V_ + v) * N_;
    bool allp = true;
    for (int n = 0; n < N_; ++n) allp &= (mp[n] != 0);
    cnt += allp ? 0 : 1;
  }
  last[b] = (cnt > 0) ? (cnt - 1) : (V_ - 1);
}

// ---------------- big GEMM: C1[12800, GP] = X[12800, 8192] @ Wkbf^T ----------------
// WG = 256 thr = 8 waves (2 M x 4 N). Wave tile 64x32 -> 8 accumulators.
// Double-buffered fragments: stage s=0/1 alternate, k unrolled by 64.
__global__ void __launch_bounds__(256)
gemm_xwk_kernel(const unsigned short* __restrict__ X,
                const unsigned short* __restrict__ Wkbf,
                float* __restrict__ C1) {
  int wave = threadIdx.x >> 5, lane = threadIdx.x & 31;
  int wm = wave >> 2, wn = wave & 3;
  int m0 = blockIdx.y * 128 + wm * 64;
  int n0 = blockIdx.x * 128 + wn * 32;
  int ml = lane & 15;

  v8f acc[4][2] = {};
  v16bf a[2][4], b[2][2];

  // prologue: stage 0 @ k=0
#pragma unroll
  for (int i = 0; i < 4; ++i) a[0][i] = load_fragA(X,    m0 + i * 16 + ml, ND_, 0, lane);
#pragma unroll
  for (int j = 0; j < 2; ++j) b[0][j] = load_fragB(Wkbf, n0 + j * 16 + ml, ND_, 0, lane);

  for (int k0 = 0; k0 < ND_; k0 += 64) {
    // prefetch stage 1 @ k0+32 (always in range: ND_ % 64 == 0)
#pragma unroll
    for (int i = 0; i < 4; ++i) a[1][i] = load_fragA(X,    m0 + i * 16 + ml, ND_, k0 + 32, lane);
#pragma unroll
    for (int j = 0; j < 2; ++j) b[1][j] = load_fragB(Wkbf, n0 + j * 16 + ml, ND_, k0 + 32, lane);
    // compute stage 0
#pragma unroll
    for (int i = 0; i < 4; ++i)
#pragma unroll
      for (int j = 0; j < 2; ++j) acc[i][j] = wmma_bf16(a[0][i], b[0][j], acc[i][j]);
    // prefetch stage 0 @ k0+64
    if (k0 + 64 < ND_) {
#pragma unroll
      for (int i = 0; i < 4; ++i) a[0][i] = load_fragA(X,    m0 + i * 16 + ml, ND_, k0 + 64, lane);
#pragma unroll
      for (int j = 0; j < 2; ++j) b[0][j] = load_fragB(Wkbf, n0 + j * 16 + ml, ND_, k0 + 64, lane);
    }
    // compute stage 1
#pragma unroll
    for (int i = 0; i < 4; ++i)
#pragma unroll
      for (int j = 0; j < 2; ++j) acc[i][j] = wmma_bf16(a[1][i], b[1][j], acc[i][j]);
  }

  int mb = (lane >> 4) * 8;
  int nc = lane & 15;
#pragma unroll
  for (int i = 0; i < 4; ++i)
#pragma unroll
    for (int j = 0; j < 2; ++j)
#pragma unroll
      for (int r = 0; r < 8; ++r)
        C1[(size_t)(m0 + i * 16 + mb + r) * GP_ + (n0 + j * 16 + nc)] = acc[i][j][r];
}

// ---------------- per-step recurrent GEMM ----------------
// xout[b,g] = C1[b*V + t, g] + bias1[g] + vt[b,t]*colI[g] + sum_k hbf[b,k]*Wrbf[g,k]
// WG = 256 thr = 8 waves (2 M x 4 N), wave tile 32x32, grid (13, 4).
// K = 384 = 6 * 64 -> double-buffered, unrolled by 64.
__global__ void __launch_bounds__(256)
gemm_hstep_kernel(const unsigned short* __restrict__ hbf,
                  const unsigned short* __restrict__ Wrbf,
                  const float* __restrict__ C1,
                  const float* __restrict__ bias1,
                  const float* __restrict__ colI,
                  const float* __restrict__ vt,
                  float* __restrict__ xout, int t) {
  int wave = threadIdx.x >> 5, lane = threadIdx.x & 31;
  int wm = wave >> 2, wn = wave & 3;
  int m0 = blockIdx.y * 64 + wm * 32;
  int n0 = blockIdx.x * 128 + wn * 32;
  int ml = lane & 15;

  v8f acc[2][2] = {};
  v16bf a[2][2], b[2][2];

#pragma unroll
  for (int i = 0; i < 2; ++i) a[0][i] = load_fragA(hbf,  m0 + i * 16 + ml, H_, 0, lane);
#pragma unroll
  for (int j = 0; j < 2; ++j) b[0][j] = load_fragB(Wrbf, n0 + j * 16 + ml, H_, 0, lane);

  for (int k0 = 0; k0 < H_; k0 += 64) {
#pragma unroll
    for (int i = 0; i < 2; ++i) a[1][i] = load_fragA(hbf,  m0 + i * 16 + ml, H_, k0 + 32, lane);
#pragma unroll
    for (int j = 0; j < 2; ++j) b[1][j] = load_fragB(Wrbf, n0 + j * 16 + ml, H_, k0 + 32, lane);
#pragma unroll
    for (int i = 0; i < 2; ++i)
#pragma unroll
      for (int j = 0; j < 2; ++j) acc[i][j] = wmma_bf16(a[0][i], b[0][j], acc[i][j]);
    if (k0 + 64 < H_) {
#pragma unroll
      for (int i = 0; i < 2; ++i) a[0][i] = load_fragA(hbf,  m0 + i * 16 + ml, H_, k0 + 64, lane);
#pragma unroll
      for (int j = 0; j < 2; ++j) b[0][j] = load_fragB(Wrbf, n0 + j * 16 + ml, H_, k0 + 64, lane);
    }
#pragma unroll
    for (int i = 0; i < 2; ++i)
#pragma unroll
      for (int j = 0; j < 2; ++j) acc[i][j] = wmma_bf16(a[1][i], b[1][j], acc[i][j]);
  }

  int mb = (lane >> 4) * 8;
  int nc = lane & 15;
#pragma unroll
  for (int i = 0; i < 2; ++i)
#pragma unroll
    for (int j = 0; j < 2; ++j)
#pragma unroll
      for (int r = 0; r < 8; ++r) {
        int m = m0 + i * 16 + mb + r;        // batch index
        int n = n0 + j * 16 + nc;            // gate index
        float v = acc[i][j][r]
                + C1[((size_t)m * V_ + t) * GP_ + n]
                + bias1[n]
                + vt[m * V_ + t] * colI[n];
        xout[(size_t)m * GP_ + n] = v;
      }
}

// ---------------- per-step gate / state / (conditional) head ----------------
// One block (128 threads) per batch element.
__global__ void __launch_bounds__(128)
gate_step_kernel(const float* __restrict__ xout,
                 float* __restrict__ cst,            // [B,H] f32 cell state
                 unsigned short* __restrict__ hbf,   // [B,H] bf16 hidden (GEMM input)
                 float* __restrict__ tmp_h,          // [KW,B,H] ring
                 float* __restrict__ tmp_dis,        // [KW,B] ring
                 const int* __restrict__ lastArr,
                 const float* __restrict__ Ws,  const float* __restrict__ bs,
                 const float* __restrict__ Wrs, const float* __restrict__ brs,
                 const float* __restrict__ Wc,  const float* __restrict__ bc,
                 const float* __restrict__ Wo,  const float* __restrict__ bo,
                 float* __restrict__ out, int t) {
  __shared__ float sfm[L_], sim[L_], sh_cur;
  __shared__ float sh_hnew[H_], sh_theme[H_], sh_s1[H_ / 6], sh_r[H_], sh_ld[KW_];
  __shared__ float sh_lh[H_ * KW_];

  int b = blockIdx.x;
  int tid = threadIdx.x;
  const float* xr = xout + (size_t)b * GP_;
  int slot = t % KW_;

  if (tid == 0) {
    // fm = cumsum(softmax(xout[:L])), im = reverse-cumsum(softmax(xout[L:2L]))
    float z[L_], p[L_];
    float mx = -1e30f, s = 0.f, cum = 0.f;
    for (int l = 0; l < L_; ++l) { z[l] = xr[l]; mx = fmaxf(mx, z[l]); }
    for (int l = 0; l < L_; ++l) { p[l] = __expf(z[l] - mx); s += p[l]; }
    for (int l = 0; l < L_; ++l) { cum += p[l] / s; sfm[l] = cum; }
    mx = -1e30f; s = 0.f; cum = 0.f;
    for (int l = 0; l < L_; ++l) { z[l] = xr[L_ + l]; mx = fmaxf(mx, z[l]); }
    for (int l = 0; l < L_; ++l) { p[l] = __expf(z[l] - mx); s += p[l]; }
    for (int l = L_ - 1; l >= 0; --l) { cum += p[l] / s; sim[l] = cum; }
    float mean = 0.f;
    for (int l = 0; l < L_; ++l) mean += sfm[l];
    sh_cur = 1.0f - mean / (float)L_;
    tmp_dis[slot * B_ + b] = sh_cur;
  }
  __syncthreads();

  // LSTM-like cell update with master gates
  for (int e = tid; e < H_; e += 128) {
    int l = e / CH_, ch = e - l * CH_;
    float gf = sigm(xr[2 * L_ + l * CH_ + ch]);
    float gi = sigm(xr[2 * L_ + (L_ + l) * CH_ + ch]);
    float go = sigm(xr[2 * L_ + (2 * L_ + l) * CH_ + ch]);
    float ci = tanhf(xr[2 * L_ + (3 * L_ + l) * CH_ + ch]);
    float cl = cst[(size_t)b * H_ + e];
    float fmv = sfm[l], imv = sim[l], ov = fmv * imv;
    float cn = ov * (gf * cl + gi * ci) + (fmv - ov) * cl + (imv - ov) * ci;
    float hn = go * tanhf(cn);
    cst[(size_t)b * H_ + e] = cn;
    hbf[(size_t)b * H_ + e] = f2bf(hn);
    sh_hnew[e] = hn;
    tmp_h[((size_t)slot * B_ + b) * H_ + e] = hn;
  }
  __syncthreads();

  int lastb = lastArr[b];
  if (lastb != t) return;   // uniform per block

  // local_dis = softmax(cumsum over window of distances)
  if (tid == 0) {
    float cv[KW_], cum = 0.f;
    for (int j = 0; j < KW_; ++j) {
      int sj = t - (KW_ - 1) + j;
      float dj = 0.f;
      if (sj >= 0) dj = (sj == t) ? sh_cur : tmp_dis[(sj % KW_) * B_ + b];
      cum += dj; cv[j] = cum;
    }
    float mx = -1e30f, s = 0.f;
    for (int j = 0; j < KW_; ++j) mx = fmaxf(mx, cv[j]);
    for (int j = 0; j < KW_; ++j) { cv[j] = __expf(cv[j] - mx); s += cv[j]; }
    for (int j = 0; j < KW_; ++j) sh_ld[j] = cv[j] / s;
  }
  __syncthreads();

  // local_h and theme (mean over window)
  for (int e = tid; e < H_; e += 128) {
    float accm = 0.f;
    for (int j = 0; j < KW_; ++j) {
      int sj = t - (KW_ - 1) + j;
      float v = 0.f;
      if (sj >= 0)
        v = (sj == t) ? sh_hnew[e] : tmp_h[((size_t)(sj % KW_) * B_ + b) * H_ + e];
      float lh = v * sh_ld[j];
      sh_lh[e * KW_ + j] = lh;
      accm += lh;
    }
    sh_theme[e] = accm * (1.0f / (float)KW_);
  }
  __syncthreads();

  // theme MLP: sigmoid(relu(theme @ Ws.T + bs) @ Wrs.T + brs)
  if (tid < H_ / 6) {
    float s = bs[tid];
    const float* w = Ws + (size_t)tid * H_;
    for (int e = 0; e < H_; ++e) s += sh_theme[e] * w[e];
    sh_s1[tid] = fmaxf(s, 0.f);
  }
  __syncthreads();

  // conv (Conv1d, out len 1) fused with theme gate and residual h
  for (int o = tid; o < H_; o += 128) {
    float th = brs[o];
    const float* wr2 = Wrs + (size_t)o * (H_ / 6);
    for (int u = 0; u < H_ / 6; ++u) th += sh_s1[u] * wr2[u];
    th = sigm(th);
    float co = bc[o];
    const float* wc = Wc + (size_t)o * H_ * KW_;
    for (int e2 = 0; e2 < H_; ++e2) {
      const float* lh = &sh_lh[e2 * KW_];
      const float* w2 = &wc[e2 * KW_];
#pragma unroll
      for (int j = 0; j < KW_; ++j) co += lh[j] * w2[j];
    }
    sh_r[o] = th * co + sh_hnew[o];
  }
  __syncthreads();

  // output projection Wo [OUT, H]
  if (tid < OUT_) {
    float s = bo[tid];
    const float* w = Wo + (size_t)tid * H_;
    for (int e = 0; e < H_; ++e) s += sh_r[e] * w[e];
    out[(size_t)b * OUT_ + tid] = s;
  }
}

// ---------------- host ----------------

extern "C" void kernel_launch(void* const* d_in, const int* in_sizes, int n_in,
                              void* d_out, int out_size, void* d_ws, size_t ws_size,
                              hipStream_t stream) {
  const int*           node_ids    = (const int*)d_in[0];
  const float*         visit_times = (const float*)d_in[3];
  const unsigned char* attn_mask   = (const unsigned char*)d_in[5];
  const float*         embed       = (const float*)d_in[6];
  const float*         Wk          = (const float*)d_in[7];
  const float*         bk          = (const float*)d_in[8];
  const float*         Wr          = (const float*)d_in[9];
  const float*         br          = (const float*)d_in[10];
  const float*         Ws          = (const float*)d_in[11];
  const float*         bs          = (const float*)d_in[12];
  const float*         Wrs         = (const float*)d_in[13];
  const float*         brs         = (const float*)d_in[14];
  const float*         Wc          = (const float*)d_in[15];
  const float*         bc          = (const float*)d_in[16];
  const float*         Wo          = (const float*)d_in[17];
  const float*         bo          = (const float*)d_in[18];
  float* out = (float*)d_out;
  char*  ws  = (char*)d_ws;

  // workspace layout (all offsets 256B-aligned)
  const size_t oXbf   = 0;                                   // 12800*8192*2
  const size_t oWkbf  = oXbf   + (size_t)12800 * ND_ * 2;    // 1664*8192*2
  const size_t oWrbf  = oWkbf  + (size_t)GP_ * ND_ * 2;      // 1664*384*2
  const size_t oC1    = oWrbf  + (size_t)GP_ * H_ * 2;       // 12800*1664*4
  const size_t oXout  = oC1    + (size_t)12800 * GP_ * 4;    // 256*1664*4
  const size_t oBias  = oXout  + (size_t)B_ * GP_ * 4;       // 1664*4
  const size_t oColI  = oBias  + (size_t)GP_ * 4;
  const size_t oLast  = oColI  + (size_t)GP_ * 4;            // 256*4
  const size_t oState = oLast  + 1024;
  const size_t oC     = oState;                              // 256*384*4
  const size_t oHbf   = oC     + (size_t)B_ * H_ * 4;        // 256*384*2
  const size_t oTmpH  = oHbf   + (size_t)B_ * H_ * 2;        // 10*256*384*4
  const size_t oTmpD  = oTmpH  + (size_t)KW_ * B_ * H_ * 4;  // 10*256*4
  const size_t stateBytes = (oTmpD + (size_t)KW_ * B_ * 4) - oState;

  unsigned short* Xbf    = (unsigned short*)(ws + oXbf);
  unsigned short* Wkbf   = (unsigned short*)(ws + oWkbf);
  unsigned short* Wrbf   = (unsigned short*)(ws + oWrbf);
  float*          C1     = (float*)(ws + oC1);
  float*          xoutB  = (float*)(ws + oXout);
  float*          bias1  = (float*)(ws + oBias);
  float*          colI   = (float*)(ws + oColI);
  int*            lastA  = (int*)(ws + oLast);
  float*          cst    = (float*)(ws + oC);
  unsigned short* hbf    = (unsigned short*)(ws + oHbf);
  float*          tmp_h  = (float*)(ws + oTmpH);
  float*          tmp_d  = (float*)(ws + oTmpD);

  // zero recurrent state (harness poisons ws)
  hipMemsetAsync(ws + oState, 0, stateBytes, stream);

  gather_x_kernel <<<12800, 256, 0, stream>>>(node_ids, embed, Xbf);
  pad_wk_kernel   <<<GP_,   256, 0, stream>>>(Wk, Wkbf);
  pad_wr_kernel   <<<GP_,   128, 0, stream>>>(Wr, Wrbf);
  prep_misc_kernel<<<13,    128, 0, stream>>>(Wk, Wr, bk, br, bias1, colI);
  last_kernel     <<<1,     256, 0, stream>>>(attn_mask, lastA);

  // hoisted big GEMM: X @ Wk.T  (M=12800, N=1664, K=8192)
  gemm_xwk_kernel<<<dim3(13, 100), 256, 0, stream>>>(Xbf, Wkbf, C1);

  // sequential scan: 50 x (recurrent GEMM + gate update)
  for (int t = 0; t < V_; ++t) {
    gemm_hstep_kernel<<<dim3(13, 4), 256, 0, stream>>>(hbf, Wrbf, C1, bias1, colI,
                                                       visit_times, xoutB, t);
    gate_step_kernel<<<B_, 128, 0, stream>>>(xoutB, cst, hbf, tmp_h, tmp_d, lastA,
                                             Ws, bs, Wrs, brs, Wc, bc, Wo, bo, out, t);
  }
}
